// RNNDecoder_46832323395601
// MI455X (gfx1250) — matrix operand order repaired and checked by reading, not compiled, not run on hardware
//
#include <hip/hip_runtime.h>
#include <hip/hip_bf16.h>

// ---------------------------------------------------------------------------
// MI455X (gfx1250) persistent RNN decoder.
//   - wave32, WMMA bf16 16x16x32, fp32 accum
//   - weights staged once into L2-resident bf16 WMMA-tile layout
//   - 16 persistent workgroups, global sense-reversing barrier per phase
// ---------------------------------------------------------------------------

typedef __bf16 bf16_t;
typedef __attribute__((ext_vector_type(16))) __bf16 v16bf;
typedef __attribute__((ext_vector_type(8)))  float  v8f;

#define ENC   128
#define HID   512
#define NLAY  3
#define KATT  10
#define BSZ   64
#define TSTEPS 512
#define TCTX  128
#define NOUT  121          // 6*MO+1
#define NWG   16
#define NTHREADS 256
#define CHUNK_KT 8         // k-tiles per LDS chunk: 8*32*64*2B = 32 KB

// K geometry (all multiples of 32)
#define K0_PAD 160         // layer0 input 131 -> 160
#define K0_TOT 672         // + 512 recurrent
#define KN_PAD 672         // layerN input 643 -> 672
#define KN_TOT 1184        // + 512 recurrent
#define KO_TOT 1536        // output concat 3*512
#define KT0 (K0_TOT/32)    // 21
#define KTN (KN_TOT/32)    // 37
#define KTO (KO_TOT/32)    // 48
#define NT_G 128           // gate n-tiles  (2048/16)
#define NT_O 8             // output n-tiles (128/16)

__device__ __forceinline__ float sigm(float x) { return 1.0f / (1.0f + __expf(-x)); }
__device__ __forceinline__ v8f vzero8() { v8f z = {0.f,0.f,0.f,0.f,0.f,0.f,0.f,0.f}; return z; }

// ---------------------------------------------------------------------------
// Global split barrier across the persistent grid (lives in L2 via d_ws).
// ---------------------------------------------------------------------------
__device__ void wg_global_barrier(unsigned* bar) {
    __syncthreads();
    if (threadIdx.x == 0) {
        volatile unsigned* vep = bar + 1;
        unsigned e = *vep;
        __threadfence();                       // release prior global writes
        unsigned arrived = atomicAdd(bar, 1u);
        if (arrived == NWG - 1) {
            atomicExch(bar, 0u);
            __threadfence();
            atomicAdd(bar + 1, 1u);            // flip epoch, releases everyone
        } else {
            while (*vep == e) { __builtin_amdgcn_s_sleep(1); }
        }
        __threadfence();                       // acquire
    }
    __syncthreads();
}

// ---------------------------------------------------------------------------
// Activation element for A-matrix staging.
// phase 0: layer0   : [x(3) | w(128) | pad->160 | h_slot0(512)]
// phase 1: layerN[0]: [x | w_new | h_slot2 | pad->672 | h_slot1(512)]
// phase 2: layerN[1]: [x | w_new | h_slot0 | pad->672 | h_slot2(512)]
// phase 4: output   : [h_slot0 | h_slot1 | h_slot2]
// ---------------------------------------------------------------------------
__device__ __forceinline__ float a_val(int phase, int b, int k, int t,
                                       const float* strokes, const float* wv,
                                       const float* hin, const float* hrec) {
    if (k < 3)   return strokes[(b * TSTEPS + t) * 3 + k];
    if (k < 131) return wv[b * ENC + (k - 3)];
    const int Kin  = (phase == 0) ? 131 : 643;
    const int Kpad = (phase == 0) ? K0_PAD : KN_PAD;
    if (k < Kin)  return hin[b * HID + (k - 131)];
    if (k < Kpad) return 0.0f;
    return hrec[b * HID + (k - Kpad)];
}

__device__ __forceinline__ float a_val_out(int b, int k, const float* h0,
                                           const float* h1, const float* h2) {
    if (k < HID)     return h0[b * HID + k];
    if (k < 2 * HID) return h1[b * HID + k - HID];
    return h2[b * HID + k - 2 * HID];
}

// Stage one K-chunk of the A matrix into LDS in the exact WMMA A-operand
// lane layout: [ktile][mtile(4)][lane(32)][elem(16)], bf16.
// A 16x32 tile: lanes 0-15 -> M=lane, K = e; lanes 16-31 same M,
// K = 16 + e (contiguous, matches B staging).
__device__ void stage_chunk(int phase, int t, int kt0, int ktc,
                            const float* strokes, const float* wv,
                            const float* hin, const float* hrec,
                            const float* h0, const float* h1, const float* h2,
                            bf16_t* sA) {
    const int total = ktc << 11;               // ktc * 4 * 32 * 16
    for (int idx = threadIdx.x; idx < total; idx += NTHREADS) {
        int e   = idx & 15;
        int ln  = (idx >> 4) & 31;
        int mt  = (idx >> 9) & 3;
        int ktl = idx >> 11;
        int kk  = ((ln >> 4) << 4) + e;
        int k = (kt0 + ktl) * 32 + kk;
        int m = mt * 16 + (ln & 15);
        float v = (phase == 4) ? a_val_out(m, k, h0, h1, h2)
                               : a_val(phase, m, k, t, strokes, wv, hin, hrec);
        sA[idx] = (bf16_t)v;
    }
}

// ---------------------------------------------------------------------------
// Gate GEMM: [64 x K] x [K x 2048] restricted to this wave's 16-wide n-tile.
// Each wave owns (gate q = wave>>1, half s = wave&1) of this WG's 32 hidden
// units; accumulates 4 M-tiles over the whole K.
// ---------------------------------------------------------------------------
__device__ void gemm_gate_phase(int phase, int t, int Kt, const bf16_t* Bst,
                                int ntile,
                                const float* strokes, const float* wv,
                                const float* hin, const float* hrec,
                                bf16_t* sA, v8f acc[4], int lane) {
    const v16bf* Bp = (const v16bf*)Bst;
    for (int kt0 = 0; kt0 < Kt; kt0 += CHUNK_KT) {
        int ktc = Kt - kt0; if (ktc > CHUNK_KT) ktc = CHUNK_KT;
        stage_chunk(phase, t, kt0, ktc, strokes, wv, hin, hrec,
                    nullptr, nullptr, nullptr, sA);
        __syncthreads();
        for (int ktl = 0; ktl < ktc; ++ktl) {
            v16bf bv = Bp[(size_t)((kt0 + ktl) * NT_G + ntile) * 32 + lane];
#pragma unroll
            for (int mt = 0; mt < 4; ++mt) {
                v16bf av = *(const v16bf*)(sA + ((((ktl << 2) + mt) * 32) + lane) * 16);
                acc[mt] = __builtin_amdgcn_wmma_f32_16x16x32_bf16(
                    false, av, false, bv, (short)0, acc[mt], false, false);
            }
        }
        __syncthreads();
    }
}

// Scatter WMMA accumulators into LDS as gates[q][b][u] (q=gate, u=local unit).
// C/D layout: VGPR r, lanes 0-15 -> M=r, lanes 16-31 -> M=r+8; N = lane&15.
__device__ __forceinline__ void store_gates(const v8f acc[4], float* sG,
                                            int lane, int wave) {
    const int u = ((wave & 1) << 4) + (lane & 15);
    const int q = wave >> 1;
#pragma unroll
    for (int mt = 0; mt < 4; ++mt) {
#pragma unroll
        for (int r = 0; r < 8; ++r) {
            int b = mt * 16 + r + ((lane >> 4) << 3);
            sG[(q * BSZ + b) * 32 + u] = acc[mt][r];
        }
    }
}

__device__ void lstm_pointwise(int wg, const float* bias, const float* c_src,
                               float* h_dst, float* c_dst, const float* sG) {
    for (int it = threadIdx.x; it < BSZ * 32; it += NTHREADS) {
        int b = it >> 5, u = it & 31;
        int ug = wg * 32 + u;
        float gi = sG[(0 * BSZ + b) * 32 + u] + bias[0 * HID + ug];
        float gf = sG[(1 * BSZ + b) * 32 + u] + bias[1 * HID + ug];
        float gg = sG[(2 * BSZ + b) * 32 + u] + bias[2 * HID + ug];
        float go = sG[(3 * BSZ + b) * 32 + u] + bias[3 * HID + ug];
        float c  = c_src[b * HID + ug];
        float c2 = sigm(gf) * c + sigm(gi) * tanhf(gg);
        float h  = sigm(go) * tanhf(c2);
        h_dst[b * HID + ug] = h;
        if (c_dst) c_dst[b * HID + ug] = c2;
    }
}

// ---------------------------------------------------------------------------
// Persistent main kernel: 16 WGs x 256 threads, loops t = 0..511 internally.
// ---------------------------------------------------------------------------
extern "C" __global__ __launch_bounds__(NTHREADS, 1)
void rnn_persistent(const float* __restrict__ strokes,
                    const float* __restrict__ context,
                    const float* __restrict__ b0,
                    const float* __restrict__ b_n,
                    const float* __restrict__ W_att,
                    const float* __restrict__ b_att,
                    const float* __restrict__ b_out,
                    const bf16_t* __restrict__ B0s, const bf16_t* __restrict__ B1s,
                    const bf16_t* __restrict__ B2s, const bf16_t* __restrict__ BOs,
                    float* hS, float* cS, float* h0buf, float* wv, float* kap,
                    unsigned* bar, float* out) {
    __shared__ __align__(32) unsigned char smem[32768];
    bf16_t* sA = (bf16_t*)smem;   // A-operand staging (32 KB)
    float*  sG = (float*)smem;    // gate exchange 4x64x32 f32 (32 KB, reused)

    const int wg = blockIdx.x;
    const int tid = threadIdx.x;
    const int lane = tid & 31;
    const int wave = tid >> 5;

    float* h0s = hS;                  float* c0s = cS;
    float* h1s = hS + BSZ * HID;      float* c1s = cS + BSZ * HID;
    float* h2s = hS + 2 * BSZ * HID;  float* c2s = cS + 2 * BSZ * HID;

    // this wave's n-tile inside the 2048-wide gate matrix:
    // gate q = wave>>1, WG owns units [wg*32, wg*32+32), half s = wave&1
    const int ntile_g = (wave >> 1) * 32 + wg * 2 + (wave & 1);

    for (int t = 0; t < TSTEPS; ++t) {
        // ---------------- P1: layer0 LSTM (output only feeds attention) ----
        {
            v8f acc[4] = {vzero8(), vzero8(), vzero8(), vzero8()};
            gemm_gate_phase(0, t, KT0, B0s, ntile_g,
                            strokes, wv, nullptr, h0s, sA, acc, lane);
            store_gates(acc, sG, lane, wave);
            __syncthreads();
            lstm_pointwise(wg, b0, c0s, h0buf, nullptr, sG);
            wg_global_barrier(bar);
        }
        // ---------------- P2: Gaussian mixture attention ------------------
        {
            float* sAtt = sG;          // [4][32]: alpha(0..9) beta(10..19) kappa(20..29)
            float* sPhi = sG + 128;    // [4][128]
            const int bb = wg * 4;     // 4 batches per WG
            for (int it = tid; it < 120; it += NTHREADS) {
                int bi = it / 30, j = it % 30, b = bb + bi;
                const float* hp = h0buf + b * HID;
                const float* wp = W_att + j * HID;
                float s = b_att[j];
                for (int u = 0; u < HID; ++u) s += hp[u] * wp[u];
                sAtt[bi * 32 + j] = __expf(s);
            }
            __syncthreads();
            for (int it = tid; it < 4 * KATT; it += NTHREADS) {
                int bi = it / KATT, j = it % KATT, b = bb + bi;
                float kn = sAtt[bi * 32 + 20 + j] + kap[b * KATT + j];
                kap[b * KATT + j] = kn;
                sAtt[bi * 32 + 20 + j] = kn;
            }
            __syncthreads();
            for (int it = tid; it < 4 * TCTX; it += NTHREADS) {
                int bi = it >> 7, u = it & 127;
                float s = 0.f;
                for (int j = 0; j < KATT; ++j) {
                    float d = sAtt[bi * 32 + 20 + j] - (float)u;
                    s += sAtt[bi * 32 + j] * __expf(-sAtt[bi * 32 + 10 + j] * d * d);
                }
                sPhi[bi * TCTX + u] = s;
            }
            __syncthreads();
            for (int it = tid; it < 4 * ENC; it += NTHREADS) {
                int bi = it >> 7, e = it & 127, b = bb + bi;
                const float* cx = context + (size_t)b * TCTX * ENC + e;
                float s = 0.f;
                for (int u = 0; u < TCTX; ++u) s += sPhi[bi * TCTX + u] * cx[u * ENC];
                wv[b * ENC + e] = s;
            }
            wg_global_barrier(bar);
        }
        // ---------------- P3: layerN[0] -> writes slot0 -------------------
        {
            v8f acc[4] = {vzero8(), vzero8(), vzero8(), vzero8()};
            gemm_gate_phase(1, t, KTN, B1s, ntile_g,
                            strokes, wv, h2s, h1s, sA, acc, lane);
            store_gates(acc, sG, lane, wave);
            __syncthreads();
            lstm_pointwise(wg, b_n, c1s, h0s, c0s, sG);
            wg_global_barrier(bar);
        }
        // ---------------- P4: layerN[1] -> writes slot1 -------------------
        {
            v8f acc[4] = {vzero8(), vzero8(), vzero8(), vzero8()};
            gemm_gate_phase(2, t, KTN, B2s, ntile_g,
                            strokes, wv, h0s, h2s, sA, acc, lane);
            store_gates(acc, sG, lane, wave);
            __syncthreads();
            lstm_pointwise(wg, b_n + 4 * HID, c2s, h1s, c1s, sG);
            wg_global_barrier(bar);
        }
        // ---------------- P5: output GEMM [64x1536]x[1536x121] ------------
        {
            const bool act = (wg < NT_O) && (wave < 4);
            v8f acc = vzero8();
            const v16bf* Bp = (const v16bf*)BOs;
            for (int kt0 = 0; kt0 < KTO; kt0 += CHUNK_KT) {
                int ktc = KTO - kt0; if (ktc > CHUNK_KT) ktc = CHUNK_KT;
                stage_chunk(4, t, kt0, ktc, nullptr, nullptr, nullptr, nullptr,
                            h0s, h1s, h2s, sA);
                __syncthreads();
                if (act) {
                    for (int ktl = 0; ktl < ktc; ++ktl) {
                        v16bf bv = Bp[(size_t)((kt0 + ktl) * NT_O + wg) * 32 + lane];
                        v16bf av = *(const v16bf*)(sA + ((((ktl << 2) + wave) * 32) + lane) * 16);
                        acc = __builtin_amdgcn_wmma_f32_16x16x32_bf16(
                            false, av, false, bv, (short)0, acc, false, false);
                    }
                }
                __syncthreads();
            }
            if (act) {
                int n = wg * 16 + (lane & 15);
                if (n < NOUT) {
                    float bo = b_out[n];
                    int bbase = wave * 16 + ((lane >> 4) << 3);
#pragma unroll
                    for (int r = 0; r < 8; ++r) {
                        int b = bbase + r;
                        out[((size_t)b * TSTEPS + t) * NOUT + n] = acc[r] + bo;
                    }
                }
            }
            wg_global_barrier(bar);
        }
    }
}

// ---------------------------------------------------------------------------
// Prep: stage fp32 weights into bf16 WMMA B-operand tile order.
// B 32x16 tile: lanes 0-15 hold K=0..15 (elem e = K), lanes 16-31 K=16..31,
// column N = lane&15. Tile stream: [ktile][ntile][lane][16].
// Combined K layout per layer: [W_ih^T rows, zero-padded | W_hh^T rows].
// ---------------------------------------------------------------------------
extern "C" __global__ void stage_b_kernel(const float* __restrict__ Wih,
                                          const float* __restrict__ Whh,
                                          bf16_t* __restrict__ dst,
                                          int Kin, int Kpad, int Khh,
                                          int NT, int Nsrc, int total) {
    int idx = blockIdx.x * blockDim.x + threadIdx.x;
    if (idx >= total) return;
    int e  = idx & 15;
    int ln = (idx >> 4) & 31;
    int nt = (idx >> 9) % NT;
    int kt = idx / (NT << 9);
    int kk = ((ln >> 4) << 4) + e;
    int k  = kt * 32 + kk;
    int n  = nt * 16 + (ln & 15);
    float v = 0.f;
    if (n < Nsrc) {
        if (k < Kpad) { if (k < Kin) v = Wih[(size_t)n * Kin + k]; }
        else          { int kh = k - Kpad; if (kh < Khh) v = Whh[(size_t)n * Khh + kh]; }
    }
    dst[idx] = (bf16_t)v;
}

extern "C" __global__ void init_state_kernel(float* hS, float* cS, float* wv,
                                             float* kap, unsigned* bar,
                                             const float* h_0, const float* c_0,
                                             const float* w_0, const float* k_0) {
    int idx = blockIdx.x * blockDim.x + threadIdx.x;
    if (idx < NLAY * BSZ * HID) {
        int l = idx / (BSZ * HID);
        int u = idx % HID;
        hS[idx] = h_0[l * HID + u];
        cS[idx] = c_0[l * HID + u];
    }
    if (idx < BSZ * ENC)  wv[idx]  = w_0[idx % ENC];
    if (idx < BSZ * KATT) kap[idx] = k_0[idx % KATT];
    if (idx < 2) bar[idx] = 0u;
}

// ---------------------------------------------------------------------------
extern "C" void kernel_launch(void* const* d_in, const int* in_sizes, int n_in,
                              void* d_out, int out_size, void* d_ws, size_t ws_size,
                              hipStream_t stream) {
    const float* strokes = (const float*)d_in[0];
    const float* context = (const float*)d_in[1];
    const float* W_ih0   = (const float*)d_in[2];
    const float* W_hh0   = (const float*)d_in[3];
    const float* b0      = (const float*)d_in[4];
    const float* W_ih_n  = (const float*)d_in[5];
    const float* W_hh_n  = (const float*)d_in[6];
    const float* b_n     = (const float*)d_in[7];
    const float* W_att   = (const float*)d_in[8];
    const float* b_att   = (const float*)d_in[9];
    const float* W_out   = (const float*)d_in[10];
    const float* b_out   = (const float*)d_in[11];
    const float* h_0     = (const float*)d_in[12];
    const float* c_0     = (const float*)d_in[13];
    const float* w_0     = (const float*)d_in[14];
    const float* k_0     = (const float*)d_in[15];
    float* out = (float*)d_out;

    // workspace carving (256B aligned)
    char* ws = (char*)d_ws;
    size_t off = 0;
    auto carve = [&](size_t bytes) {
        void* p = ws + off;
        off = (off + bytes + 255) & ~(size_t)255;
        return p;
    };
    const int nB0 = KT0 * NT_G * 512;   // 21*128*512
    const int nBN = KTN * NT_G * 512;   // 37*128*512
    const int nBO = KTO * NT_O * 512;   // 48*8*512
    bf16_t* B0s = (bf16_t*)carve((size_t)nB0 * 2);
    bf16_t* B1s = (bf16_t*)carve((size_t)nBN * 2);
    bf16_t* B2s = (bf16_t*)carve((size_t)nBN * 2);
    bf16_t* BOs = (bf16_t*)carve((size_t)nBO * 2);
    float* hS    = (float*)carve((size_t)NLAY * BSZ * HID * 4);
    float* cS    = (float*)carve((size_t)NLAY * BSZ * HID * 4);
    float* h0buf = (float*)carve((size_t)BSZ * HID * 4);
    float* wv    = (float*)carve((size_t)BSZ * ENC * 4);
    float* kap   = (float*)carve((size_t)BSZ * KATT * 4);
    unsigned* bar = (unsigned*)carve(256);

    // stage weights (bf16, WMMA B tile order)
    stage_b_kernel<<<(nB0 + 255) / 256, 256, 0, stream>>>(
        W_ih0, W_hh0, B0s, 131, K0_PAD, HID, NT_G, 2048, nB0);
    stage_b_kernel<<<(nBN + 255) / 256, 256, 0, stream>>>(
        W_ih_n, W_hh_n, B1s, 643, KN_PAD, HID, NT_G, 2048, nBN);
    stage_b_kernel<<<(nBN + 255) / 256, 256, 0, stream>>>(
        W_ih_n + (size_t)2048 * 643, W_hh_n + (size_t)2048 * 512, B2s,
        643, KN_PAD, HID, NT_G, 2048, nBN);
    stage_b_kernel<<<(nBO + 255) / 256, 256, 0, stream>>>(
        W_out, W_out, BOs, KO_TOT, KO_TOT, 0, NT_O, NOUT, nBO);

    init_state_kernel<<<(NLAY * BSZ * HID + 255) / 256, 256, 0, stream>>>(
        hS, cS, wv, kap, bar, h_0, c_0, w_0, k_0);

    rnn_persistent<<<NWG, NTHREADS, 0, stream>>>(
        strokes, context, b0, b_n, W_att, b_att, b_out,
        B0s, B1s, B2s, BOs, hS, cS, h0buf, wv, kap, bar, out);
}
